// Attention_55130200211640
// MI455X (gfx1250) — matrix operand order repaired and checked by reading, not compiled, not run on hardware
//
#include <hip/hip_runtime.h>
#include <stdint.h>

// ---------------- types ----------------
typedef __bf16  bf16_t;
typedef bf16_t  bf16x16 __attribute__((ext_vector_type(16)));
typedef float   f32x8   __attribute__((ext_vector_type(8)));

// ---------------- problem constants ----------------
constexpr int Bc   = 4;
constexpr int Sc   = 2048;
constexpr int HIDc = 2048;
constexpr int Hc   = 16;
constexpr int KVc  = 4;
constexpr int Dc   = 128;
constexpr int TOK  = Bc * Sc;            // 8192 tokens
constexpr float SCALE = 0.08838834764831845f;  // 1/sqrt(128)

// ---------------- scalar helpers ----------------
__device__ __forceinline__ uint16_t f32_to_bf16(float f) {
  uint32_t u = __builtin_bit_cast(uint32_t, f);
  u += 0x7FFFu + ((u >> 16) & 1u);       // round-to-nearest-even
  return (uint16_t)(u >> 16);
}
__device__ __forceinline__ float bf16_to_f32(uint16_t h) {
  uint32_t u = ((uint32_t)h) << 16;
  return __builtin_bit_cast(float, u);
}
__device__ __forceinline__ f32x8 wmma_bf16(bf16x16 a, bf16x16 b, f32x8 c) {
  // D = A(16x32 bf16) * B(32x16 bf16) + C(16x16 f32)
  return __builtin_amdgcn_wmma_f32_16x16x32_bf16(false, a, false, b, (short)0, c,
                                                 false, false);
}

// ---------------- CDNA5 async global->LDS (ASYNCcnt) ----------------
// GVS addressing: mem = SADDR(u64) + VADDR(i32) + inst_offset; VDST vgpr = LDS byte addr.
__device__ __forceinline__ void async_b128(uint32_t lds_off, const void* sbase,
                                           uint32_t voff) {
  asm volatile("global_load_async_to_lds_b128 %0, %1, %2"
               :: "v"(lds_off), "v"(voff), "s"(sbase) : "memory");
}
__device__ __forceinline__ void wait_async0() {
  asm volatile("s_wait_asynccnt 0" ::: "memory");
}
// Low 32 bits of a flat pointer into the LDS aperture are the LDS byte offset.
__device__ __forceinline__ uint32_t lds_off_of(const void* p) {
  return (uint32_t)(uintptr_t)p;
}

// ---------------- kernel 1: f32 -> bf16 convert ----------------
__global__ void cvt_f32_bf16(const float* __restrict__ in,
                             uint16_t* __restrict__ out, int n) {
  int i = (blockIdx.x * blockDim.x + threadIdx.x) * 4;
  if (i >= n) return;
  float4 v = *(const float4*)(in + i);
  ushort4 o;
  o.x = f32_to_bf16(v.x); o.y = f32_to_bf16(v.y);
  o.z = f32_to_bf16(v.z); o.w = f32_to_bf16(v.w);
  *(ushort4*)(out + i) = o;
}

// ---------------- kernel 2: generic WMMA GEMM  C = A(MxK) * B(NxK)^T ----------------
// A,B bf16 row-major with K contiguous. Block: 256 thr (8 waves), tile 128x128x32.
constexpr int BM = 128, BN = 128, BK = 32;
constexpr int LDT = 40;   // LDS row stride in elems (80B -> conflict-free, 16B aligned)

template <bool OUT_F32>
__global__ __launch_bounds__(256) void gemm_bf16_wmma(
    const uint16_t* __restrict__ A, const uint16_t* __restrict__ B,
    void* __restrict__ C, int M, int N, int K) {
  __shared__ uint16_t sA[2][BM * LDT];
  __shared__ uint16_t sB[2][BN * LDT];

  const int tid  = threadIdx.x;
  const int lane = tid & 31, wave = tid >> 5;
  const int half = lane >> 4, l16 = lane & 15;
  const int wm = wave >> 2;          // 0..1 -> 64 rows each
  const int wn = wave & 3;           // 0..3 -> 32 cols each
  const int bm0 = blockIdx.y * BM;
  const int bn0 = blockIdx.x * BN;

  f32x8 acc[4][2];
#pragma unroll
  for (int i = 0; i < 4; ++i)
#pragma unroll
    for (int j = 0; j < 2; ++j)
#pragma unroll
      for (int r = 0; r < 8; ++r) acc[i][j][r] = 0.f;

  auto stage = [&](int buf, int kt) {
#pragma unroll
    for (int s = 0; s < 2; ++s) {            // A tile: 128 rows x 64B
      int idx = tid + s * 256;
      int row = idx >> 2, seg = idx & 3;
      async_b128(lds_off_of(&sA[buf][row * LDT + seg * 8]), A,
                 (uint32_t)((((size_t)(bm0 + row)) * K + kt * BK + seg * 8) * 2));
    }
#pragma unroll
    for (int s = 0; s < 2; ++s) {            // B tile: 128 rows x 64B
      int idx = tid + s * 256;
      int row = idx >> 2, seg = idx & 3;
      async_b128(lds_off_of(&sB[buf][row * LDT + seg * 8]), B,
                 (uint32_t)((((size_t)(bn0 + row)) * K + kt * BK + seg * 8) * 2));
    }
  };

  const int nk = K / BK;
  stage(0, 0);
  for (int kt = 0; kt < nk; ++kt) {
    wait_async0();
    __syncthreads();
    if (kt + 1 < nk) stage((kt + 1) & 1, kt + 1);
    const int buf = kt & 1;

    bf16x16 af[4], bfr[2];
#pragma unroll
    for (int i = 0; i < 4; ++i) {  // A frag: K 0-7/16-23 (lanes 0-15) | 8-15/24-31
      const uint16_t* p = &sA[buf][(wm * 64 + i * 16 + l16) * LDT + 8 * half];
      ((uint4*)&af[i])[0] = *(const uint4*)p;
      ((uint4*)&af[i])[1] = *(const uint4*)(p + 16);
    }
#pragma unroll
    for (int j = 0; j < 2; ++j) {  // B frag: K 0-7/16-23 (lanes 0-15) | 8-15/24-31
      const uint16_t* p = &sB[buf][(wn * 32 + j * 16 + l16) * LDT + 8 * half];
      ((uint4*)&bfr[j])[0] = *(const uint4*)p;
      ((uint4*)&bfr[j])[1] = *(const uint4*)(p + 16);
    }
#pragma unroll
    for (int i = 0; i < 4; ++i)
#pragma unroll
      for (int j = 0; j < 2; ++j) acc[i][j] = wmma_bf16(af[i], bfr[j], acc[i][j]);
  }

#pragma unroll
  for (int i = 0; i < 4; ++i)
#pragma unroll
    for (int j = 0; j < 2; ++j)
#pragma unroll
      for (int r = 0; r < 8; ++r) {
        size_t m = bm0 + wm * 64 + i * 16 + r + 8 * half;
        size_t n = bn0 + wn * 32 + j * 16 + l16;
        float v = acc[i][j][r];
        if (OUT_F32) ((float*)C)[m * N + n] = v;
        else         ((uint16_t*)C)[m * N + n] = f32_to_bf16(v);
      }
}

// ---------------- kernel 3: RoPE (in place, bf16) ----------------
__global__ void rope_kernel(uint16_t* __restrict__ qk, const float* __restrict__ cosb,
                            const float* __restrict__ sinb, int heads) {
  int idx = blockIdx.x * blockDim.x + threadIdx.x;      // (token, head, dpair)
  int total = TOK * heads * (Dc / 2);
  if (idx >= total) return;
  int d   = idx % 64;
  int h   = (idx / 64) % heads;
  int tok = idx / (64 * heads);
  int s   = tok % Sc;
  uint16_t* base = qk + (size_t)tok * heads * Dc + h * Dc;
  float lo = bf16_to_f32(base[d]);
  float hi = bf16_to_f32(base[d + 64]);
  float c0 = cosb[s * Dc + d],      s0 = sinb[s * Dc + d];
  float c1 = cosb[s * Dc + d + 64], s1 = sinb[s * Dc + d + 64];
  base[d]      = f32_to_bf16(lo * c0 - hi * s0);   // rot_half: -t[d+64]
  base[d + 64] = f32_to_bf16(hi * c1 + lo * s1);   // rot_half:  t[d]
}

// ---------------- kernel 4: V transpose  [tok][kv*128] -> [b][kv][d][s] ----------------
__global__ void transpose_v(const uint16_t* __restrict__ V, uint16_t* __restrict__ Vt) {
  int idx = blockIdx.x * blockDim.x + threadIdx.x;
  if (idx >= TOK * KVc * Dc) return;
  int d   = idx % Dc;
  int kv  = (idx / Dc) % KVc;
  int tok = idx / (KVc * Dc);
  int b = tok / Sc, s = tok % Sc;
  Vt[(((size_t)b * KVc + kv) * Dc + d) * Sc + s] = V[idx];
}

// ---------------- kernel 5: flash attention (double-buffered async staging) ----------------
// Block = (128 queries, 1 head, 1 batch); 8 waves x 16 rows. KV tiles of 64 keys.
constexpr int KT = 64;          // keys per tile
constexpr int KP = Dc + 8;      // sK row stride (elems): key-major [key][d]
constexpr int VP = KT + 8;      // sV row stride: d-major [d][key]
constexpr int PP = KT + 8;      // per-wave P stride: [m][key]

__global__ __launch_bounds__(256) void flash_attn(
    const uint16_t* __restrict__ Q,   // [tok][H*D] (roped)
    const uint16_t* __restrict__ Kx,  // [tok][KV*D] (roped)
    const uint16_t* __restrict__ Vt,  // [B][KV][D][S]
    uint16_t* __restrict__ O) {       // [tok][H*D]
  __shared__ uint16_t sK[2][KT * KP];     // 2 x 17408 B
  __shared__ uint16_t sV[2][Dc * VP];     // 2 x 18432 B
  __shared__ uint16_t sP[8][16 * PP];     // 18432 B   (total ~90 KB, fits WGP LDS)

  const int tid  = threadIdx.x;
  const int lane = tid & 31, wave = tid >> 5;
  const int half = lane >> 4, l16 = lane & 15;
  const int qt = blockIdx.x, h = blockIdx.y, b = blockIdx.z;
  const int kv = h >> 2;                      // GQA: rep = 4
  const int q0 = qt * 128 + wave * 16;

  // Preload this wave's Q strip as 4 A-fragments (D = 4 x 32).
  bf16x16 qf[4];
  {
    const size_t qrow = ((size_t)(b * Sc + q0 + l16)) * (Hc * Dc) + h * Dc;
#pragma unroll
    for (int ks = 0; ks < 4; ++ks) {
      const uint16_t* p = Q + qrow + ks * 32 + 8 * half;
      ((uint4*)&qf[ks])[0] = *(const uint4*)p;
      ((uint4*)&qf[ks])[1] = *(const uint4*)(p + 16);
    }
  }

  f32x8 oacc[8];
  float rmax[8], rsum[8];
#pragma unroll
  for (int i = 0; i < 8; ++i) {
#pragma unroll
    for (int r = 0; r < 8; ++r) oacc[i][r] = 0.f;
    rmax[i] = -__builtin_inff();
    rsum[i] = 0.f;
  }

  auto stage = [&](int buf, int j) {
#pragma unroll
    for (int s = 0; s < 4; ++s) {             // K tile: 64 rows x 256B
      int idx = tid + s * 256;
      int row = idx >> 4, seg = idx & 15;
      async_b128(lds_off_of(&sK[buf][row * KP + seg * 8]), Kx,
                 (uint32_t)((((size_t)(b * Sc + j * KT + row)) * (KVc * Dc) +
                             kv * Dc + seg * 8) * 2));
    }
#pragma unroll
    for (int s = 0; s < 4; ++s) {             // V^T tile: 128 rows x 128B
      int idx = tid + s * 256;
      int row = idx >> 3, seg = idx & 7;
      async_b128(lds_off_of(&sV[buf][row * VP + seg * 8]), Vt,
                 (uint32_t)(((((size_t)(b * KVc + kv)) * Dc + row) * Sc +
                             j * KT + seg * 8) * 2));
    }
  };

  constexpr int NT = Sc / KT;                 // 32 KV tiles
  stage(0, 0);
  for (int j = 0; j < NT; ++j) {
    wait_async0();
    __syncthreads();
    if (j + 1 < NT) stage((j + 1) & 1, j + 1);   // overlap copy with compute
    const int buf = j & 1;

    // ---- scores: 16 x 64 strip per wave ----
    f32x8 sacc[4];
#pragma unroll
    for (int nt = 0; nt < 4; ++nt) {
#pragma unroll
      for (int r = 0; r < 8; ++r) sacc[nt][r] = 0.f;
#pragma unroll
      for (int ks = 0; ks < 4; ++ks) {
        bf16x16 kb;
        const uint16_t* p = &sK[buf][(nt * 16 + l16) * KP + ks * 32 + 8 * half];
        ((uint4*)&kb)[0] = *(const uint4*)p;
        ((uint4*)&kb)[1] = *(const uint4*)(p + 16);
        sacc[nt] = wmma_bf16(qf[ks], kb, sacc[nt]);
      }
    }

    // ---- online softmax (row = r + 8*half; reductions stay in 16-lane group) ----
    float lsum[8], alpha[8];
#pragma unroll
    for (int r = 0; r < 8; ++r) {
      float m = sacc[0][r];
#pragma unroll
      for (int nt = 1; nt < 4; ++nt) m = fmaxf(m, sacc[nt][r]);
#pragma unroll
      for (int off = 1; off < 16; off <<= 1) m = fmaxf(m, __shfl_xor(m, off, 32));
      float nm = fmaxf(rmax[r], m);
      alpha[r] = __expf((rmax[r] - nm) * SCALE);
      rmax[r] = nm;
      lsum[r] = 0.f;
    }
#pragma unroll
    for (int nt = 0; nt < 4; ++nt)
#pragma unroll
      for (int r = 0; r < 8; ++r) {
        float p = __expf((sacc[nt][r] - rmax[r]) * SCALE);
        sacc[nt][r] = p;
        lsum[r] += p;
      }
#pragma unroll
    for (int r = 0; r < 8; ++r) {
#pragma unroll
      for (int off = 1; off < 16; off <<= 1) lsum[r] += __shfl_xor(lsum[r], off, 32);
      rsum[r] = rsum[r] * alpha[r] + lsum[r];
    }
#pragma unroll
    for (int nt = 0; nt < 8; ++nt)
#pragma unroll
      for (int r = 0; r < 8; ++r) oacc[nt][r] *= alpha[r];

    // ---- P: C-layout -> A-layout via per-wave LDS (same-wave, DScnt ordered) ----
#pragma unroll
    for (int nt = 0; nt < 4; ++nt)
#pragma unroll
      for (int r = 0; r < 8; ++r)
        sP[wave][(r + 8 * half) * PP + nt * 16 + l16] = f32_to_bf16(sacc[nt][r]);

    bf16x16 pf[2];
#pragma unroll
    for (int ks = 0; ks < 2; ++ks) {
      const uint16_t* p = &sP[wave][l16 * PP + ks * 32 + 8 * half];
      ((uint4*)&pf[ks])[0] = *(const uint4*)p;
      ((uint4*)&pf[ks])[1] = *(const uint4*)(p + 16);
    }

    // ---- O += P * V ----
#pragma unroll
    for (int nt = 0; nt < 8; ++nt)
#pragma unroll
      for (int ks = 0; ks < 2; ++ks) {
        bf16x16 vb;
        const uint16_t* p = &sV[buf][(nt * 16 + l16) * VP + ks * 32 + 8 * half];
        ((uint4*)&vb)[0] = *(const uint4*)p;
        ((uint4*)&vb)[1] = *(const uint4*)(p + 16);
        oacc[nt] = wmma_bf16(pf[ks], vb, oacc[nt]);
      }
    // no trailing barrier: next iteration's wait+barrier orders buffer reuse
  }

  // ---- finalize ----
#pragma unroll
  for (int r = 0; r < 8; ++r) {
    float inv = 1.0f / rsum[r];
    size_t row = ((size_t)(b * Sc + q0 + r + 8 * half)) * (Hc * Dc) + h * Dc;
#pragma unroll
    for (int nt = 0; nt < 8; ++nt)
      O[row + nt * 16 + l16] = f32_to_bf16(oacc[nt][r] * inv);
  }
}

// ---------------- host: launch pipeline ----------------
extern "C" void kernel_launch(void* const* d_in, const int* in_sizes, int n_in,
                              void* d_out, int out_size, void* d_ws, size_t ws_size,
                              hipStream_t stream) {
  (void)in_sizes; (void)n_in; (void)out_size; (void)ws_size;
  const float* x    = (const float*)d_in[0];
  const float* cosb = (const float*)d_in[1];
  const float* sinb = (const float*)d_in[2];
  const float* wq   = (const float*)d_in[3];
  const float* wk   = (const float*)d_in[4];
  const float* wv   = (const float*)d_in[5];
  const float* wo   = (const float*)d_in[6];

  uint8_t* ws = (uint8_t*)d_ws;
  size_t off = 0;
  auto carve = [&](size_t bytes) {
    void* p = ws + off;
    off += (bytes + 255) & ~(size_t)255;
    return p;
  };
  uint16_t* xb  = (uint16_t*)carve((size_t)TOK * HIDc * 2);
  uint16_t* wqb = (uint16_t*)carve((size_t)Hc * Dc * HIDc * 2);
  uint16_t* wkb = (uint16_t*)carve((size_t)KVc * Dc * HIDc * 2);
  uint16_t* wvb = (uint16_t*)carve((size_t)KVc * Dc * HIDc * 2);
  uint16_t* wob = (uint16_t*)carve((size_t)HIDc * Hc * Dc * 2);
  uint16_t* Qb  = (uint16_t*)carve((size_t)TOK * Hc * Dc * 2);
  uint16_t* Kb  = (uint16_t*)carve((size_t)TOK * KVc * Dc * 2);
  uint16_t* Vb  = (uint16_t*)carve((size_t)TOK * KVc * Dc * 2);
  uint16_t* Vtb = (uint16_t*)carve((size_t)TOK * KVc * Dc * 2);
  uint16_t* Ob  = (uint16_t*)carve((size_t)TOK * Hc * Dc * 2);

  auto nb4 = [](long long n) { return (unsigned)((n / 4 + 255) / 256); };
  cvt_f32_bf16<<<nb4((long long)TOK * HIDc), 256, 0, stream>>>(x, xb, TOK * HIDc);
  cvt_f32_bf16<<<nb4((long long)Hc * Dc * HIDc), 256, 0, stream>>>(wq, wqb, Hc * Dc * HIDc);
  cvt_f32_bf16<<<nb4((long long)KVc * Dc * HIDc), 256, 0, stream>>>(wk, wkb, KVc * Dc * HIDc);
  cvt_f32_bf16<<<nb4((long long)KVc * Dc * HIDc), 256, 0, stream>>>(wv, wvb, KVc * Dc * HIDc);
  cvt_f32_bf16<<<nb4((long long)HIDc * Hc * Dc), 256, 0, stream>>>(wo, wob, HIDc * Hc * Dc);

  gemm_bf16_wmma<false><<<dim3(Hc * Dc / BN, TOK / BM), 256, 0, stream>>>(
      xb, wqb, Qb, TOK, Hc * Dc, HIDc);
  gemm_bf16_wmma<false><<<dim3(KVc * Dc / BN, TOK / BM), 256, 0, stream>>>(
      xb, wkb, Kb, TOK, KVc * Dc, HIDc);
  gemm_bf16_wmma<false><<<dim3(KVc * Dc / BN, TOK / BM), 256, 0, stream>>>(
      xb, wvb, Vb, TOK, KVc * Dc, HIDc);

  rope_kernel<<<(TOK * Hc * 64 + 255) / 256, 256, 0, stream>>>(Qb, cosb, sinb, Hc);
  rope_kernel<<<(TOK * KVc * 64 + 255) / 256, 256, 0, stream>>>(Kb, cosb, sinb, KVc);
  transpose_v<<<(TOK * KVc * Dc + 255) / 256, 256, 0, stream>>>(Vb, Vtb);

  flash_attn<<<dim3(Sc / 128, Hc, Bc), 256, 0, stream>>>(Qb, Kb, Vtb, Ob);

  gemm_bf16_wmma<true><<<dim3(HIDc / BN, TOK / BM), 256, 0, stream>>>(
      Ob, wob, d_out, TOK, HIDc, Hc * Dc);
}
